// LabelPropagator_56951266345702
// MI455X (gfx1250) — matrix operand order repaired and checked by reading, not compiled, not run on hardware
//
#include <hip/hip_runtime.h>
#include <hip/hip_bf16.h>
#include <math.h>

typedef __attribute__((ext_vector_type(16))) _Float16 v16h;
typedef __attribute__((ext_vector_type(8)))  _Float16 v8h;
typedef __attribute__((ext_vector_type(8)))  float    v8f;

#define WMMA_F32_F16(a, b, c) \
  __builtin_amdgcn_wmma_f32_16x16x32_f16(false, (a), false, (b), (short)0, (c), false, false)

// Problem constants (from setup_inputs)
constexpr int S  = 2;
constexpr int C  = 256;    // feature dim
constexpr int KD = 128;    // key dim
constexpr int CE = 16;     // pos-enc dim (dv of cross attention)
constexpr int HW = 2304;   // 48*48
constexpr int LQ = 2304;   // query length
constexpr int LK = 11520;  // F*h*w memory length
constexpr int NS = 4;      // key-dimension split factor (occupancy)
constexpr float INV_TAU = 30.0f;

// ---------------- workspace layout (bytes) ----------------
constexpr size_t OFF_WQK = 0;                                  // f16 [S][LQ][KD]
constexpr size_t SZ_WQK  = (size_t)S * LQ * KD * 2;
constexpr size_t OFF_WVS = OFF_WQK + SZ_WQK;                   // f16 [S][C][LQ]  (transposed V)
constexpr size_t SZ_WVS  = (size_t)S * LQ * C * 2;
constexpr size_t OFF_WKC = OFF_WVS + SZ_WVS;                   // f16 [S][LK][KD]
constexpr size_t SZ_WKC  = (size_t)S * LK * KD * 2;
constexpr size_t OFF_WVC = OFF_WKC + SZ_WKC;                   // f16 [S][CE][LK] (transposed V)
constexpr size_t SZ_WVC  = (size_t)S * LK * CE * 2;
constexpr size_t OFF_WQC = OFF_WVC + SZ_WVC;                   // f16 [S][LQ][KD]
constexpr size_t SZ_WQC  = (size_t)S * LQ * KD * 2;
constexpr size_t OFF_ATT = OFF_WQC + SZ_WQC;                   // f32 [S][LQ][C]
constexpr size_t SZ_ATT  = (size_t)S * LQ * C * 4;
constexpr size_t OFF_XN  = OFF_ATT + SZ_ATT;                   // f32 [S][LQ][C]
constexpr size_t SZ_XN   = (size_t)S * LQ * C * 4;
constexpr size_t OFF_MS  = OFF_XN + SZ_XN;                     // f32 [S*LQ] x4 (final m/l self, m/l cross)
constexpr size_t SZ_ML   = (size_t)S * LQ * 4;
constexpr size_t OFF_LS  = OFF_MS + SZ_ML;
constexpr size_t OFF_MC  = OFF_LS + SZ_ML;
constexpr size_t OFF_LC  = OFF_MC + SZ_ML;
constexpr size_t OFF_PMS = OFF_LC + SZ_ML;                     // f32 [NS][S*LQ] partial m (self)
constexpr size_t SZ_PML  = (size_t)NS * S * LQ * 4;
constexpr size_t OFF_PLS = OFF_PMS + SZ_PML;                   // partial l (self)
constexpr size_t OFF_PMC = OFF_PLS + SZ_PML;                   // partial m (cross)
constexpr size_t OFF_PLC = OFF_PMC + SZ_PML;                   // partial l (cross)
constexpr size_t OFF_PO  = OFF_PLC + SZ_PML;                   // f32 [NS][S*CE*HW] partial cross out
constexpr size_t SZ_PO   = (size_t)NS * S * CE * HW * 4;

// =========================================================================
// proj_qk: Y = l2norm_rows(X @ W^T + b) -> f16 [S][M][KD]
// layout 0: X channel-major [F][S][C][HW]; layout 1: row-major [S][M][C]
// One wave computes a full 16-row x 128-col block (8 N tiles), so the row
// L2 norm can be done locally with half-wave shuffles.
// =========================================================================
__global__ __launch_bounds__(256) void proj_qk_kernel(
    const float* __restrict__ X, const float* __restrict__ W,
    const float* __restrict__ bias, _Float16* __restrict__ Y, int M, int layout) {
  int wid = blockIdx.x * (blockDim.x >> 5) + (threadIdx.x >> 5);
  int nwaves = (M >> 4) * S;
  if (wid >= nwaves) return;
  int s = wid % S, rb = wid / S, q0 = rb << 4;
  int lane = threadIdx.x & 31;
  int rowq = q0 + (lane & 15);
  int kA = (lane >> 4) * 16;
  int kB = (lane >> 4) * 16;

  size_t base; size_t chs;
  if (layout == 0) {
    int f = rowq / HW, p = rowq % HW;
    base = (size_t)f * ((size_t)S * C * HW) + (size_t)(s * C) * HW + p;
    chs = HW;
  } else {
    base = ((size_t)s * M + rowq) * C;
    chs = 1;
  }

  v16h a[8];
#pragma unroll
  for (int kt = 0; kt < 8; ++kt) {
    int k0 = kt * 32 + kA;
#pragma unroll
    for (int e = 0; e < 8; ++e) {
      a[kt][e]     = (_Float16)X[base + (size_t)(k0 + e) * chs];
      a[kt][8 + e] = (_Float16)X[base + (size_t)(k0 + 8 + e) * chs];
    }
  }

  v8f acc[8];
#pragma unroll
  for (int nt = 0; nt < 8; ++nt) {
    int wrow = nt * 16 + (lane & 15);
    const float* wp = W + (size_t)wrow * C;
    v8f c = {};
#pragma unroll
    for (int kt = 0; kt < 8; ++kt) {
      v16h b;
      const float* bp = wp + kt * 32 + kB;
#pragma unroll
      for (int e = 0; e < 16; ++e) b[e] = (_Float16)bp[e];
      c = WMMA_F32_F16(a[kt], b, c);
    }
    float bv = bias[wrow];
#pragma unroll
    for (int r = 0; r < 8; ++r) c[r] += bv;
    acc[nt] = c;
  }

  float rn[8];
#pragma unroll
  for (int r = 0; r < 8; ++r) {
    float ss = 0.f;
#pragma unroll
    for (int nt = 0; nt < 8; ++nt) ss += acc[nt][r] * acc[nt][r];
    ss += __shfl_xor(ss, 8, 32);
    ss += __shfl_xor(ss, 4, 32);
    ss += __shfl_xor(ss, 2, 32);
    ss += __shfl_xor(ss, 1, 32);
    rn[r] = 1.0f / fmaxf(sqrtf(ss), 1e-12f);
  }
  int ro = (lane >> 4) * 8;
#pragma unroll
  for (int nt = 0; nt < 8; ++nt)
#pragma unroll
    for (int r = 0; r < 8; ++r) {
      int q = q0 + r + ro;
      Y[((size_t)s * M + q) * KD + nt * 16 + (lane & 15)] = (_Float16)(acc[nt][r] * rn[r]);
    }
}

// =========================================================================
// wv_self projection: Yt = (X @ W^T + b)^T -> f16 [S][C][LQ] (transposed so
// the PV B-fragment reads contiguous keys). One wave = 16x16 tile; each lane
// emits one contiguous 8-half (16B) vector store.
// =========================================================================
__global__ __launch_bounds__(256) void proj_v_self_kernel(
    const float* __restrict__ X, const float* __restrict__ W,
    const float* __restrict__ bias, _Float16* __restrict__ Yt) {
  int wid = blockIdx.x * (blockDim.x >> 5) + (threadIdx.x >> 5);
  int nwaves = (LQ >> 4) * 16 * S;
  if (wid >= nwaves) return;
  int s = wid % S, t = wid / S;
  int nt = t % 16, rb = t / 16, q0 = rb << 4;
  int lane = threadIdx.x & 31;
  int rowq = q0 + (lane & 15);
  int kA = (lane >> 4) * 16, kB = (lane >> 4) * 16;
  size_t base = (size_t)(s * C) * HW + rowq;  // tgt channel-major, F=1
  int wrow = nt * 16 + (lane & 15);
  const float* wp = W + (size_t)wrow * C;
  v8f c = {};
#pragma unroll
  for (int kt = 0; kt < 8; ++kt) {
    v16h a, b;
    int k0 = kt * 32;
#pragma unroll
    for (int e = 0; e < 8; ++e) {
      a[e]     = (_Float16)X[base + (size_t)(k0 + kA + e) * HW];
      a[8 + e] = (_Float16)X[base + (size_t)(k0 + kA + 8 + e) * HW];
    }
    const float* bp = wp + k0 + kB;
#pragma unroll
    for (int e = 0; e < 16; ++e) b[e] = (_Float16)bp[e];
    c = WMMA_F32_F16(a, b, c);
  }
  float bv = bias[wrow];
  int ro = (lane >> 4) * 8;
  v8h vec;
#pragma unroll
  for (int r = 0; r < 8; ++r) vec[r] = (_Float16)(c[r] + bv);
  // transposed store: column wrow, rows q0+ro .. q0+ro+7 contiguous
  *(v8h*)(Yt + ((size_t)(s * C) + wrow) * LQ + q0 + ro) = vec;
}

// =========================================================================
// wv_cross projection (tiny K=16 GEMM) -> f16 [S][CE][LK] transposed.
// Thread index == output index (k fastest) so both the pos-enc reads and the
// output writes are coalesced.
// =========================================================================
__global__ void proj_vcross_kernel(const float* __restrict__ P, const float* __restrict__ W,
                                   const float* __restrict__ bias, _Float16* __restrict__ Yt) {
  int idx = blockIdx.x * blockDim.x + threadIdx.x;
  int total = S * CE * LK;
  if (idx >= total) return;
  int k = idx % LK, t = idx / LK;
  int d = t % CE, s = t / CE;
  int f = k / HW, p = k % HW;
  size_t base = (size_t)f * ((size_t)S * CE * HW) + (size_t)(s * CE) * HW + p;
  float acc = bias[d];
  const float* wr = W + d * CE;
#pragma unroll
  for (int j = 0; j < CE; ++j) acc += P[base + (size_t)j * HW] * wr[j];
  Yt[idx] = (_Float16)acc;
}

// =========================================================================
// Softmax stats with key-split: wave = (slice, qblock, s). Two passes over
// its key slice (max, then sum-exp) with deferred cross-lane reduction.
// Writes partial PM/PL[slice][s*Lq + q].
// =========================================================================
__global__ __launch_bounds__(256) void attn_stats_kernel(
    const _Float16* __restrict__ Q, const _Float16* __restrict__ K,
    float* __restrict__ PM, float* __restrict__ PL, int Lq, int Lk, int ns) {
  int wid = blockIdx.x * (blockDim.x >> 5) + (threadIdx.x >> 5);
  int inner = (Lq >> 4) * S;
  if (wid >= inner * ns) return;
  int sl = wid / inner, w2 = wid % inner;
  int s = w2 % S, rb = w2 / S, q0 = rb << 4;
  int klen = Lk / ns, kstart = sl * klen;
  int lane = threadIdx.x & 31;
  int kA = (lane >> 4) * 16, kB = (lane >> 4) * 16;

  v16h a[4];
  const _Float16* qp = Q + ((size_t)s * Lq + q0 + (lane & 15)) * KD;
#pragma unroll
  for (int kt = 0; kt < 4; ++kt) {
    const _Float16* src = qp + kt * 32 + kA;
#pragma unroll
    for (int e = 0; e < 8; ++e) { a[kt][e] = src[e]; a[kt][8 + e] = src[8 + e]; }
  }
  const _Float16* krow0 = K + (size_t)s * Lk * KD + (size_t)(lane & 15) * KD;

  float mrow[8];
#pragma unroll
  for (int r = 0; r < 8; ++r) mrow[r] = -1e30f;
  for (int kb = kstart; kb < kstart + klen; kb += 16) {
    const _Float16* kp = krow0 + (size_t)kb * KD;
    v8f c = {};
#pragma unroll
    for (int kt = 0; kt < 4; ++kt) {
      v16h b;
      const _Float16* src = kp + kt * 32 + kB;
#pragma unroll
      for (int e = 0; e < 16; ++e) b[e] = src[e];
      c = WMMA_F32_F16(a[kt], b, c);
    }
#pragma unroll
    for (int r = 0; r < 8; ++r) mrow[r] = fmaxf(mrow[r], c[r] * INV_TAU);
  }
#pragma unroll
  for (int r = 0; r < 8; ++r) {
    mrow[r] = fmaxf(mrow[r], __shfl_xor(mrow[r], 8, 32));
    mrow[r] = fmaxf(mrow[r], __shfl_xor(mrow[r], 4, 32));
    mrow[r] = fmaxf(mrow[r], __shfl_xor(mrow[r], 2, 32));
    mrow[r] = fmaxf(mrow[r], __shfl_xor(mrow[r], 1, 32));
  }

  float lrow[8];
#pragma unroll
  for (int r = 0; r < 8; ++r) lrow[r] = 0.f;
  for (int kb = kstart; kb < kstart + klen; kb += 16) {
    const _Float16* kp = krow0 + (size_t)kb * KD;
    v8f c = {};
#pragma unroll
    for (int kt = 0; kt < 4; ++kt) {
      v16h b;
      const _Float16* src = kp + kt * 32 + kB;
#pragma unroll
      for (int e = 0; e < 16; ++e) b[e] = src[e];
      c = WMMA_F32_F16(a[kt], b, c);
    }
#pragma unroll
    for (int r = 0; r < 8; ++r) lrow[r] += __expf(c[r] * INV_TAU - mrow[r]);
  }
#pragma unroll
  for (int r = 0; r < 8; ++r) {
    lrow[r] += __shfl_xor(lrow[r], 8, 32);
    lrow[r] += __shfl_xor(lrow[r], 4, 32);
    lrow[r] += __shfl_xor(lrow[r], 2, 32);
    lrow[r] += __shfl_xor(lrow[r], 1, 32);
  }
  int ro = (lane >> 4) * 8;
  size_t sloff = (size_t)sl * (S * Lq);
#pragma unroll
  for (int r = 0; r < 8; ++r) {
    if ((lane & 15) == r) {
      int q = q0 + r + ro;
      PM[sloff + (size_t)s * Lq + q] = mrow[r];
      PL[sloff + (size_t)s * Lq + q] = lrow[r];
    }
  }
}

// Combine per-slice (m, l) partials: m = max m_i; l = sum l_i * exp(m_i - m)
__global__ void combine_stats_kernel(const float* __restrict__ PM, const float* __restrict__ PL,
                                     float* __restrict__ M, float* __restrict__ L, int n, int ns) {
  int idx = blockIdx.x * blockDim.x + threadIdx.x;
  if (idx >= n) return;
  float m = -1e30f;
  for (int sl = 0; sl < ns; ++sl) m = fmaxf(m, PM[(size_t)sl * n + idx]);
  float l = 0.f;
  for (int sl = 0; sl < ns; ++sl)
    l += PL[(size_t)sl * n + idx] * __expf(PM[(size_t)sl * n + idx] - m);
  M[idx] = m;
  L[idx] = l;
}

// =========================================================================
// Self attention output: wave = (qblock, s, 64-dim chunk). Recompute scores,
// P (f16) staged through per-wave LDS to re-layout C-frag -> A-frag, then
// WMMA P x V into 4 accumulators. V is transposed [S][C][LQ] so B-fragment
// reads are contiguous. O = f32 [S][LQ][C].
// =========================================================================
__global__ __launch_bounds__(256) void self_out_kernel(
    const _Float16* __restrict__ Q, const _Float16* __restrict__ Km,
    const _Float16* __restrict__ Vt, const float* __restrict__ Mi,
    const float* __restrict__ Li, float* __restrict__ O) {
  __shared__ _Float16 pbuf[8][16 * 32];
  int w = threadIdx.x >> 5;
  int wid = blockIdx.x * (blockDim.x >> 5) + w;
  int nwaves = (LQ >> 4) * S * 4;
  if (wid >= nwaves) return;
  int dc = wid % 4, t = wid / 4;
  int s = t % S, rb = t / S, q0 = rb << 4;
  int lane = threadIdx.x & 31;
  int kA = (lane >> 4) * 16, kB = (lane >> 4) * 16, ro = (lane >> 4) * 8;

  v16h a[4];
  const _Float16* qp = Q + ((size_t)s * LQ + q0 + (lane & 15)) * KD;
#pragma unroll
  for (int kt = 0; kt < 4; ++kt) {
    const _Float16* src = qp + kt * 32 + kA;
#pragma unroll
    for (int e = 0; e < 8; ++e) { a[kt][e] = src[e]; a[kt][8 + e] = src[8 + e]; }
  }
  float mrow[8], linv[8];
#pragma unroll
  for (int r = 0; r < 8; ++r) {
    int q = q0 + r + ro;
    mrow[r] = Mi[(size_t)s * LQ + q];
    linv[r] = 1.0f / Li[(size_t)s * LQ + q];
  }
  v8f acc[4];
#pragma unroll
  for (int n = 0; n < 4; ++n) acc[n] = (v8f){};

  _Float16* pb = pbuf[w];
  const _Float16* krow0 = Km + (size_t)s * LQ * KD + (size_t)(lane & 15) * KD;

  for (int kb = 0; kb < LQ; kb += 32) {
#pragma unroll
    for (int t2 = 0; t2 < 2; ++t2) {
      const _Float16* kp = krow0 + (size_t)(kb + t2 * 16) * KD;
      v8f c = {};
#pragma unroll
      for (int kt = 0; kt < 4; ++kt) {
        v16h b;
        const _Float16* src = kp + kt * 32 + kB;
#pragma unroll
        for (int e = 0; e < 16; ++e) b[e] = src[e];
        c = WMMA_F32_F16(a[kt], b, c);
      }
#pragma unroll
      for (int r = 0; r < 8; ++r) {
        float p = __expf(c[r] * INV_TAU - mrow[r]) * linv[r];
        pb[(r + ro) * 32 + t2 * 16 + (lane & 15)] = (_Float16)p;
      }
    }
    // re-layout P through LDS into an A fragment (16 rows x 32 keys)
    v16h pa;
    const _Float16* pp = pb + (lane & 15) * 32 + kA;
#pragma unroll
    for (int e = 0; e < 8; ++e) { pa[e] = pp[e]; pa[8 + e] = pp[8 + e]; }
#pragma unroll
    for (int n = 0; n < 4; ++n) {
      v16h vb;
      const _Float16* vp = Vt + ((size_t)(s * C) + dc * 64 + n * 16 + (lane & 15)) * LQ + kb + kB;
#pragma unroll
      for (int e = 0; e < 16; ++e) vb[e] = vp[e];
      acc[n] = WMMA_F32_F16(pa, vb, acc[n]);
    }
  }
#pragma unroll
  for (int n = 0; n < 4; ++n)
#pragma unroll
    for (int r = 0; r < 8; ++r) {
      int q = q0 + r + ro;
      O[((size_t)s * LQ + q) * C + dc * 64 + n * 16 + (lane & 15)] = acc[n][r];
    }
}

// =========================================================================
// InstanceNorm2d (no affine): one block per (s, channel); x = tgt + attn.
// =========================================================================
__global__ __launch_bounds__(256) void instnorm_kernel(
    const float* __restrict__ T, const float* __restrict__ A, float* __restrict__ X) {
  __shared__ float red[256];
  int s = blockIdx.x / C;
  int ch = blockIdx.x % C;
  const float* tp = T + ((size_t)(s * C) + ch) * HW;
  const float* ap = A + (size_t)s * LQ * C + ch;
  float sum = 0.f, sq = 0.f;
  for (int q = threadIdx.x; q < HW; q += blockDim.x) {
    float v = tp[q] + ap[(size_t)q * C];
    sum += v; sq += v * v;
  }
  red[threadIdx.x] = sum; __syncthreads();
  for (int st = 128; st > 0; st >>= 1) {
    if ((int)threadIdx.x < st) red[threadIdx.x] += red[threadIdx.x + st];
    __syncthreads();
  }
  float mu = red[0] / (float)HW; __syncthreads();
  red[threadIdx.x] = sq; __syncthreads();
  for (int st = 128; st > 0; st >>= 1) {
    if ((int)threadIdx.x < st) red[threadIdx.x] += red[threadIdx.x + st];
    __syncthreads();
  }
  float var = red[0] / (float)HW - mu * mu;
  float rs = rsqrtf(var + 1e-5f);
  float* xp = X + (size_t)s * LQ * C + ch;
  for (int q = threadIdx.x; q < HW; q += blockDim.x) {
    float v = tp[q] + ap[(size_t)q * C];
    xp[(size_t)q * C] = (v - mu) * rs;
  }
}

// =========================================================================
// Cross attention output (dv=16), key-split: wave = (slice, qblock, s).
// Probabilities use the FINAL (m, l) so slice partial outputs just sum.
// Writes PO[slice][(s*16 + d)*HW + q]; reduce kernel sums slices -> d_out.
// =========================================================================
__global__ __launch_bounds__(256) void cross_out_kernel(
    const _Float16* __restrict__ Q, const _Float16* __restrict__ Kc,
    const _Float16* __restrict__ Vt, const float* __restrict__ Mi,
    const float* __restrict__ Li, float* __restrict__ PO, int ns) {
  __shared__ _Float16 pbuf[8][16 * 32];
  int w = threadIdx.x >> 5;
  int wid = blockIdx.x * (blockDim.x >> 5) + w;
  int inner = (LQ >> 4) * S;
  if (wid >= inner * ns) return;
  int sl = wid / inner, w2 = wid % inner;
  int s = w2 % S, rb = w2 / S, q0 = rb << 4;
  int klen = LK / ns, kstart = sl * klen;
  int lane = threadIdx.x & 31;
  int kA = (lane >> 4) * 16, kB = (lane >> 4) * 16, ro = (lane >> 4) * 8;

  v16h a[4];
  const _Float16* qp = Q + ((size_t)s * LQ + q0 + (lane & 15)) * KD;
#pragma unroll
  for (int kt = 0; kt < 4; ++kt) {
    const _Float16* src = qp + kt * 32 + kA;
#pragma unroll
    for (int e = 0; e < 8; ++e) { a[kt][e] = src[e]; a[kt][8 + e] = src[8 + e]; }
  }
  float mrow[8], linv[8];
#pragma unroll
  for (int r = 0; r < 8; ++r) {
    int q = q0 + r + ro;
    mrow[r] = Mi[(size_t)s * LQ + q];
    linv[r] = 1.0f / Li[(size_t)s * LQ + q];
  }
  v8f acc = {};
  _Float16* pb = pbuf[w];
  const _Float16* krow0 = Kc + (size_t)s * LK * KD + (size_t)(lane & 15) * KD;

  for (int kb = kstart; kb < kstart + klen; kb += 32) {
#pragma unroll
    for (int t2 = 0; t2 < 2; ++t2) {
      const _Float16* kp = krow0 + (size_t)(kb + t2 * 16) * KD;
      v8f c = {};
#pragma unroll
      for (int kt = 0; kt < 4; ++kt) {
        v16h b;
        const _Float16* src = kp + kt * 32 + kB;
#pragma unroll
        for (int e = 0; e < 16; ++e) b[e] = src[e];
        c = WMMA_F32_F16(a[kt], b, c);
      }
#pragma unroll
      for (int r = 0; r < 8; ++r) {
        float p = __expf(c[r] * INV_TAU - mrow[r]) * linv[r];
        pb[(r + ro) * 32 + t2 * 16 + (lane & 15)] = (_Float16)p;
      }
    }
    v16h pa;
    const _Float16* pp = pb + (lane & 15) * 32 + kA;
#pragma unroll
    for (int e = 0; e < 8; ++e) { pa[e] = pp[e]; pa[8 + e] = pp[8 + e]; }
    v16h vb;
    const _Float16* vp = Vt + ((size_t)(s * CE) + (lane & 15)) * LK + kb + kB;
#pragma unroll
    for (int e = 0; e < 16; ++e) vb[e] = vp[e];
    acc = WMMA_F32_F16(pa, vb, acc);
  }
  float* po = PO + (size_t)sl * (S * CE * HW);
#pragma unroll
  for (int r = 0; r < 8; ++r) {
    int q = q0 + r + ro;
    po[((size_t)(s * CE) + (lane & 15)) * HW + q] = acc[r];
  }
}

// Deterministic fixed-order sum of the cross-output slices -> d_out
__global__ void reduce_out_kernel(const float* __restrict__ PO, float* __restrict__ Out,
                                  int n, int ns) {
  int idx = blockIdx.x * blockDim.x + threadIdx.x;
  if (idx >= n) return;
  float sum = 0.f;
  for (int sl = 0; sl < ns; ++sl) sum += PO[(size_t)sl * n + idx];
  Out[idx] = sum;
}

// =========================================================================
extern "C" void kernel_launch(void* const* d_in, const int* in_sizes, int n_in,
                              void* d_out, int out_size, void* d_ws, size_t ws_size,
                              hipStream_t stream) {
  (void)in_sizes; (void)n_in; (void)out_size; (void)ws_size;
  const float* tgt     = (const float*)d_in[0];
  const float* memory  = (const float*)d_in[1];
  const float* pos_enc = (const float*)d_in[2];
  const float* WKs_w   = (const float*)d_in[3];
  const float* WKs_b   = (const float*)d_in[4];
  const float* WVs_w   = (const float*)d_in[5];
  const float* WVs_b   = (const float*)d_in[6];
  const float* WKc_w   = (const float*)d_in[7];
  const float* WKc_b   = (const float*)d_in[8];
  const float* WVc_w   = (const float*)d_in[9];
  const float* WVc_b   = (const float*)d_in[10];
  float* out = (float*)d_out;

  char* ws = (char*)d_ws;
  _Float16* wqk_self = (_Float16*)(ws + OFF_WQK);
  _Float16* wv_self  = (_Float16*)(ws + OFF_WVS);   // [S][C][LQ] transposed
  _Float16* wk_cross = (_Float16*)(ws + OFF_WKC);
  _Float16* wv_cross = (_Float16*)(ws + OFF_WVC);   // [S][CE][LK] transposed
  _Float16* wq_cross = (_Float16*)(ws + OFF_WQC);
  float* attn_self = (float*)(ws + OFF_ATT);
  float* x_norm    = (float*)(ws + OFF_XN);
  float* m_self    = (float*)(ws + OFF_MS);
  float* l_self    = (float*)(ws + OFF_LS);
  float* m_cross   = (float*)(ws + OFF_MC);
  float* l_cross   = (float*)(ws + OFF_LC);
  float* pm_self   = (float*)(ws + OFF_PMS);
  float* pl_self   = (float*)(ws + OFF_PLS);
  float* pm_cross  = (float*)(ws + OFF_PMC);
  float* pl_cross  = (float*)(ws + OFF_PLC);
  float* p_out     = (float*)(ws + OFF_PO);

  // Projections (all WMMA except the tiny K=16 pos-enc projection)
  proj_qk_kernel<<<36, 256, 0, stream>>>(tgt, WKs_w, WKs_b, wqk_self, LQ, 0);
  proj_v_self_kernel<<<576, 256, 0, stream>>>(tgt, WVs_w, WVs_b, wv_self);
  proj_qk_kernel<<<180, 256, 0, stream>>>(memory, WKc_w, WKc_b, wk_cross, LK, 0);
  proj_vcross_kernel<<<1440, 256, 0, stream>>>(pos_enc, WVc_w, WVc_b, wv_cross);

  // Self attention (q == k): split-key stats, combine, then output
  attn_stats_kernel<<<144, 256, 0, stream>>>(wqk_self, wqk_self, pm_self, pl_self, LQ, LQ, NS);
  combine_stats_kernel<<<18, 256, 0, stream>>>(pm_self, pl_self, m_self, l_self, S * LQ, NS);
  self_out_kernel<<<144, 256, 0, stream>>>(wqk_self, wqk_self, wv_self, m_self, l_self, attn_self);

  // Residual + InstanceNorm
  instnorm_kernel<<<512, 256, 0, stream>>>(tgt, attn_self, x_norm);

  // Cross attention: split-key stats + split-key output, deterministic reduce
  proj_qk_kernel<<<36, 256, 0, stream>>>(x_norm, WKc_w, WKc_b, wq_cross, LQ, 1);
  attn_stats_kernel<<<144, 256, 0, stream>>>(wq_cross, wk_cross, pm_cross, pl_cross, LQ, LK, NS);
  combine_stats_kernel<<<18, 256, 0, stream>>>(pm_cross, pl_cross, m_cross, l_cross, S * LQ, NS);
  cross_out_kernel<<<144, 256, 0, stream>>>(wq_cross, wk_cross, wv_cross, m_cross, l_cross, p_out, NS);
  reduce_out_kernel<<<288, 256, 0, stream>>>(p_out, out, S * CE * HW, NS);
}
